// GptOssSparseMoeBlock_52536039965400
// MI455X (gfx1250) — matrix operand order repaired and checked by reading, not compiled, not run on hardware
//
#include <hip/hip_runtime.h>
#include <math.h>

// ---------------- problem constants ----------------
#define TOK   1024
#define HD    2880
#define ID    2880
#define NE    16
#define TOPK  4
#define ALPHA 1.702f
#define LIMIT 7.0f

typedef __bf16 bf16_t;
typedef __attribute__((ext_vector_type(16))) __bf16 v16bf;
typedef __attribute__((ext_vector_type(8)))  __bf16 v8bf;
typedef __attribute__((ext_vector_type(8)))  float  v8f;
typedef __attribute__((ext_vector_type(4)))  float  v4f;

// ---------------- WMMA fragment helpers ----------------
// A (16x32 bf16, MxK): lane L holds row M=L&15; element j holds
// K = j + 16*(L>=16)  -> two contiguous 8-elem chunks.
static __device__ __forceinline__ v16bf load_a_frag(const bf16_t* rowk, int gsel) {
  const v8bf lo = *(const v8bf*)(rowk + 16 * gsel);
  const v8bf hi = *(const v8bf*)(rowk + 16 * gsel + 8);
  v16bf a;
#pragma unroll
  for (int i = 0; i < 8; ++i) { a[i] = lo[i]; a[8 + i] = hi[i]; }
  return a;
}

// B (32x16 bf16, KxN): lane L holds col N=L&15; element j holds K = j + 16*(L>=16)
// -> 16 contiguous source values. Source is fp32; convert in-register to bf16.
static __device__ __forceinline__ v16bf load_b_frag_f32(const float* p) {
  v16bf b;
#pragma unroll
  for (int i = 0; i < 4; ++i) {
    v4f f = *(const v4f*)(p + 4 * i);
#pragma unroll
    for (int j = 0; j < 4; ++j) b[4 * i + j] = (__bf16)f[j];
  }
  return b;
}

static __device__ __forceinline__ v8f wmma_bf16(v16bf a, v16bf b, v8f c) {
  // D = A*B + C, f32 accumulate  -> v_wmma_f32_16x16x32_bf16
  return __builtin_amdgcn_wmma_f32_16x16x32_bf16(false, a, false, b, (short)0, c,
                                                 false, false);
}

// fast 1/x via v_rcp_f32 (avoids IEEE div_scale/fixup chain in hot epilogue)
static __device__ __forceinline__ float fast_rcp(float x) {
  return __builtin_amdgcn_rcpf(x);
}

// ---------------- kernel 1: router + x -> bf16 ----------------
__global__ void __launch_bounds__(128)
moe_router(const float* __restrict__ x, const float* __restrict__ rw,
           const float* __restrict__ rb, int* __restrict__ counts,
           int* __restrict__ slots, int* __restrict__ eids,
           float* __restrict__ wslot, bf16_t* __restrict__ xb) {
  const int t   = blockIdx.x;
  const int tid = threadIdx.x;
  const float* xrow = x + (size_t)t * HD;

  // convert this token's hidden row to bf16
  for (int h = tid; h < HD; h += 128) xb[(size_t)t * HD + h] = (bf16_t)xrow[h];

  __shared__ float red[128];
  __shared__ float lg[NE];
  for (int e = 0; e < NE; ++e) {
    float p = 0.f;
    const float* wrow = rw + (size_t)e * HD;
    for (int h = tid; h < HD; h += 128) p += xrow[h] * wrow[h];
    red[tid] = p;
    __syncthreads();
    for (int s = 64; s > 0; s >>= 1) {
      if (tid < s) red[tid] += red[tid + s];
      __syncthreads();
    }
    if (tid == 0) lg[e] = red[0] + rb[e];
    __syncthreads();
  }

  if (tid == 0) {
    float mx = lg[0];
    for (int e = 1; e < NE; ++e) mx = fmaxf(mx, lg[e]);
    float pe[NE], se = 0.f;
    for (int e = 0; e < NE; ++e) { pe[e] = __expf(lg[e] - mx); se += pe[e]; }
    const float inv = fast_rcp(se);
    for (int e = 0; e < NE; ++e) pe[e] *= inv;

    int   idx[TOPK];
    float val[TOPK];
    float wsum = 0.f;
    for (int k = 0; k < TOPK; ++k) {
      int   bj = 0;
      float bv = pe[0];
      for (int e = 1; e < NE; ++e)
        if (pe[e] > bv) { bv = pe[e]; bj = e; }
      idx[k] = bj; val[k] = bv; pe[bj] = -1.f; wsum += bv;
    }
    const float rinv = fast_rcp(wsum);
    for (int k = 0; k < TOPK; ++k) {
      const int   e = idx[k];
      const float m = val[k] * rinv;
      const int   p = atomicAdd(&counts[e], 1);
      slots[e * TOK + p]  = t * TOPK + k;   // h-row slot for this (token, rank)
      eids[t * TOPK + k]  = e;
      wslot[t * TOPK + k] = m;
    }
  }
}

// ---------------- kernel 2: gate/up grouped GEMM + SwiGLU ----------------
// grid = (ID/64, TOK/64, NE), block = 128 (4 waves).
// Column-stripe tiling: wave w owns cols [n0+16w, n0+16w+16) for ALL 64 rows.
// -> fp32 weight fragments (loads + f32->bf16 cvt) are UNIQUE per wave;
//    only cheap bf16 activation fragments are re-read across waves (WGP$ hits).
__global__ void __launch_bounds__(128)
moe_gateup(const bf16_t* __restrict__ xb, const float* __restrict__ w1,
           const float* __restrict__ b1, const float* __restrict__ w3,
           const float* __restrict__ b3, const int* __restrict__ counts,
           const int* __restrict__ slots, bf16_t* __restrict__ hbuf) {
  const int e   = blockIdx.z;
  const int cnt = counts[e];
  const int mb  = blockIdx.y * 64;
  if (mb >= cnt) return;                       // block-uniform exit (EXEC all-1 kept)
  const int wave = threadIdx.x >> 5;
  const int lane = threadIdx.x & 31;
  const int c0   = blockIdx.x * 64 + wave * 16;  // this wave's 16 output cols
  const int N    = lane & 15;
  const int gsel = lane >> 4;

  // 4 row-subtile A base pointers (rows mb+16r+..; pad rows clamped, loads only)
  const bf16_t* abase[4];
#pragma unroll
  for (int r = 0; r < 4; ++r) {
    int arow = mb + 16 * r + N;
    if (arow >= cnt) arow = cnt - 1;
    abase[r] = xb + (size_t)(slots[e * TOK + arow] >> 2) * HD;
  }

  const float* w1row = w1 + (size_t)e * ID * HD + (size_t)(c0 + N) * HD + 16 * gsel;
  const float* w3row = w3 + (size_t)e * ID * HD + (size_t)(c0 + N) * HD + 16 * gsel;

  const v8f z = {0.f, 0.f, 0.f, 0.f, 0.f, 0.f, 0.f, 0.f};
  v8f g[4] = {z, z, z, z};
  v8f u[4] = {z, z, z, z};

  for (int k0 = 0; k0 < HD; k0 += 32) {
    v16bf a[4];
#pragma unroll
    for (int r = 0; r < 4; ++r) a[r] = load_a_frag(abase[r] + k0, gsel);
    const v16bf bg = load_b_frag_f32(w1row + k0);
#pragma unroll
    for (int r = 0; r < 4; ++r) g[r] = wmma_bf16(a[r], bg, g[r]);
    const v16bf bu = load_b_frag_f32(w3row + k0);
#pragma unroll
    for (int r = 0; r < 4; ++r) u[r] = wmma_bf16(a[r], bu, u[r]);
  }

  // C/D layout: lane L, vgpr rr -> row M = rr + 8*(L>=16), col N = L&15
  const int   col = c0 + N;
  const float bb1 = b1[e * ID + col];
  const float bb3 = b3[e * ID + col];
#pragma unroll
  for (int r = 0; r < 4; ++r) {
#pragma unroll
    for (int rr = 0; rr < 8; ++rr) {
      const int row = mb + 16 * r + rr + 8 * gsel;
      if (row < cnt) {
        const int   slot = slots[e * TOK + row];
        float gate = g[r][rr] + bb1;
        float up   = u[r][rr] + bb3;
        gate = fminf(gate, LIMIT);
        up   = fminf(fmaxf(up, -LIMIT), LIMIT);
        // act = gate * sigmoid(ALPHA*gate), fast rcp instead of IEEE divide
        const float act = gate * fast_rcp(1.f + __expf(-ALPHA * gate));
        const float hv  = (up + 1.f) * act;
        hbuf[(size_t)slot * ID + col] = (bf16_t)hv;
      }
    }
  }
}

// ---------------- kernel 3: down grouped GEMM -> per-slot partials ----------------
// Same column-stripe tiling as kernel 2.
__global__ void __launch_bounds__(128)
moe_down(const bf16_t* __restrict__ hbuf, const float* __restrict__ w2,
         const int* __restrict__ counts, const int* __restrict__ slots,
         float* __restrict__ outp) {
  const int e   = blockIdx.z;
  const int cnt = counts[e];
  const int mb  = blockIdx.y * 64;
  if (mb >= cnt) return;
  const int wave = threadIdx.x >> 5;
  const int lane = threadIdx.x & 31;
  const int c0   = blockIdx.x * 64 + wave * 16;
  const int N    = lane & 15;
  const int gsel = lane >> 4;

  const bf16_t* abase[4];
#pragma unroll
  for (int r = 0; r < 4; ++r) {
    int arow = mb + 16 * r + N;
    if (arow >= cnt) arow = cnt - 1;
    abase[r] = hbuf + (size_t)slots[e * TOK + arow] * ID;
  }

  const float* w2row = w2 + (size_t)e * HD * ID + (size_t)(c0 + N) * ID + 16 * gsel;

  const v8f z = {0.f, 0.f, 0.f, 0.f, 0.f, 0.f, 0.f, 0.f};
  v8f acc[4] = {z, z, z, z};

  for (int k0 = 0; k0 < ID; k0 += 32) {
    v16bf a[4];
#pragma unroll
    for (int r = 0; r < 4; ++r) a[r] = load_a_frag(abase[r] + k0, gsel);
    const v16bf b = load_b_frag_f32(w2row + k0);
#pragma unroll
    for (int r = 0; r < 4; ++r) acc[r] = wmma_bf16(a[r], b, acc[r]);
  }

  const int col = c0 + N;
#pragma unroll
  for (int r = 0; r < 4; ++r) {
#pragma unroll
    for (int rr = 0; rr < 8; ++rr) {
      const int row = mb + 16 * r + rr + 8 * gsel;
      if (row < cnt) {
        const int slot = slots[e * TOK + row];
        outp[(size_t)slot * HD + col] = acc[r][rr];
      }
    }
  }
}

// ---------------- kernel 4: deterministic weighted combine ----------------
__global__ void __launch_bounds__(256)
moe_combine(const float* __restrict__ outp, const float* __restrict__ b2,
            const int* __restrict__ eids, const float* __restrict__ wslot,
            float* __restrict__ y) {
  const int idx = blockIdx.x * 256 + threadIdx.x;
  if (idx >= TOK * HD) return;
  const int t = idx / HD;
  const int c = idx - t * HD;
  float acc = 0.f;
#pragma unroll
  for (int k = 0; k < TOPK; ++k) {
    const int slot = t * TOPK + k;
    const int e    = eids[slot];
    acc += wslot[slot] * (outp[(size_t)slot * HD + c] + b2[(size_t)e * HD + c]);
  }
  y[idx] = acc;
}

// ---------------- host launcher ----------------
extern "C" void kernel_launch(void* const* d_in, const int* in_sizes, int n_in,
                              void* d_out, int out_size, void* d_ws, size_t ws_size,
                              hipStream_t stream) {
  const float* x  = (const float*)d_in[0];
  const float* rw = (const float*)d_in[1];
  const float* rb = (const float*)d_in[2];
  const float* w1 = (const float*)d_in[3];
  const float* b1 = (const float*)d_in[4];
  const float* w3 = (const float*)d_in[5];
  const float* b3 = (const float*)d_in[6];
  const float* w2 = (const float*)d_in[7];
  const float* b2 = (const float*)d_in[8];
  float* y = (float*)d_out;

  // workspace layout (all 256B-aligned); total ~77 MB
  constexpr size_t OFF_COUNTS = 0;
  constexpr size_t OFF_SLOTS  = 256;
  constexpr size_t OFF_EIDS   = OFF_SLOTS + (size_t)NE * TOK * 4;
  constexpr size_t OFF_WSLOT  = OFF_EIDS + (size_t)TOK * TOPK * 4;
  constexpr size_t OFF_XB     = (OFF_WSLOT + (size_t)TOK * TOPK * 4 + 255) & ~(size_t)255;
  constexpr size_t OFF_HBUF   = (OFF_XB + (size_t)TOK * HD * 2 + 255) & ~(size_t)255;
  constexpr size_t OFF_OUTP   = (OFF_HBUF + (size_t)TOK * TOPK * ID * 2 + 255) & ~(size_t)255;

  char*   ws     = (char*)d_ws;
  int*    counts = (int*)(ws + OFF_COUNTS);
  int*    slots  = (int*)(ws + OFF_SLOTS);
  int*    eids   = (int*)(ws + OFF_EIDS);
  float*  wslot  = (float*)(ws + OFF_WSLOT);
  bf16_t* xb     = (bf16_t*)(ws + OFF_XB);
  bf16_t* hbuf   = (bf16_t*)(ws + OFF_HBUF);
  float*  outp   = (float*)(ws + OFF_OUTP);

  hipMemsetAsync(counts, 0, NE * sizeof(int), stream);

  moe_router<<<TOK, 128, 0, stream>>>(x, rw, rb, counts, slots, eids, wslot, xb);

  dim3 g2(ID / 64, TOK / 64, NE);
  moe_gateup<<<g2, 128, 0, stream>>>(xb, w1, b1, w3, b3, counts, slots, hbuf);

  dim3 g3(HD / 64, TOK / 64, NE);
  moe_down<<<g3, 128, 0, stream>>>(hbuf, w2, counts, slots, outp);

  moe_combine<<<(TOK * HD + 255) / 256, 256, 0, stream>>>(outp, b2, eids, wslot, y);
}
